// Downsampler_25022479466878
// MI455X (gfx1250) — matrix operand order repaired and checked by reading, not compiled, not run on hardware
//
#include <hip/hip_runtime.h>
#include <stdint.h>

#define T_DIM 4096
#define B_DIM 8
#define C_DIM 512
#define S_DIM 2048

#define KPAD 40   // bf16 elements per LDS row: 32 data + 8 pad (80B row, 16B-aligned reads)

typedef __attribute__((ext_vector_type(16))) __bf16 v16bf;
typedef __attribute__((ext_vector_type(8)))  __bf16 v8bf;
typedef __attribute__((ext_vector_type(8)))  float  v8f;

// Assemble a 16-element bf16 WMMA fragment from two 8-element (16B) LDS reads.
__device__ __forceinline__ v16bf ld_frag(const unsigned short* p0, const unsigned short* p1) {
    v8bf lo = *reinterpret_cast<const v8bf*>(p0);
    v8bf hi = *reinterpret_cast<const v8bf*>(p1);
    return __builtin_shufflevector(lo, hi, 0,1,2,3,4,5,6,7,8,9,10,11,12,13,14,15);
}

// Split 4 k-rows x 4 cols of fp32 into truncated-hi / residual-lo bf16 and store
// transposed into LDS ([col][k] layout). 1x v_perm_b32 packs each bf16 pair.
__device__ __forceinline__ void split_store(const float4 q[4],
                                            unsigned short* __restrict__ sh,
                                            unsigned short* __restrict__ sl,
                                            int sq, int tq) {
    float v[4][4];
    #pragma unroll
    for (int rr = 0; rr < 4; ++rr) {
        v[rr][0] = q[rr].x; v[rr][1] = q[rr].y; v[rr][2] = q[rr].z; v[rr][3] = q[rr].w;
    }
    #pragma unroll
    for (int i = 0; i < 4; ++i) {
        unsigned int u[4], lu[4];
        #pragma unroll
        for (int rr = 0; rr < 4; ++rr) {
            u[rr] = __builtin_bit_cast(unsigned int, v[rr][i]);
            float hf = __builtin_bit_cast(float, u[rr] & 0xFFFF0000u); // hi = trunc bf16
            lu[rr] = __builtin_bit_cast(unsigned int, v[rr][i] - hf);  // exact residual
        }
        // {hi16(b), hi16(a)} in one v_perm_b32
        uint2 ph = make_uint2(__builtin_amdgcn_perm(u[1], u[0], 0x07060302u),
                              __builtin_amdgcn_perm(u[3], u[2], 0x07060302u));
        uint2 pl = make_uint2(__builtin_amdgcn_perm(lu[1], lu[0], 0x07060302u),
                              __builtin_amdgcn_perm(lu[3], lu[2], 0x07060302u));
        *reinterpret_cast<uint2*>(&sh[(sq + i) * KPAD + tq]) = ph;
        *reinterpret_cast<uint2*>(&sl[(sq + i) * KPAD + tq]) = pl;
    }
}

// D[s,c] (per batch) = sum_t mask[b,t,s] * x[t,b,c], fp32 via 3-term bf16 split WMMA.
__global__ __launch_bounds__(256)
void downsample_gemm_kernel(const float* __restrict__ x,
                            const float* __restrict__ mask,
                            float* __restrict__ out) {
    __shared__ unsigned short sAh[128 * KPAD];   // mask^T hi : A[s][k]
    __shared__ unsigned short sAl[128 * KPAD];   // mask^T lo
    __shared__ unsigned short sBh[128 * KPAD];   // x^T hi    : Bt[c][k]
    __shared__ unsigned short sBl[128 * KPAD];   // x^T lo

    const int b  = blockIdx.y;
    const int s0 = (blockIdx.x >> 2) * 128;      // 16 s-blocks
    const int c0 = (blockIdx.x & 3) * 128;       // 4 c-blocks

    const int tid  = threadIdx.x;
    const int lane = tid & 31;
    const int wave = tid >> 5;
    const int w0   = wave & 3;                   // s-tile pair selector (0..3)
    const int w1   = wave >> 2;                  // c-tile quad selector (0..1)
    const int r    = lane & 15;
    const int hi16 = lane >> 4;

    // staging mapping: lane -> 4 consecutive s/c columns, wave -> 4 consecutive k rows
    const int sq = lane * 4;                     // 0..124
    const int tq = wave * 4;                     // 0..28

    v8f acc[2][4] = {};

    const float* mask_b = mask + ((size_t)b * T_DIM) * S_DIM + (size_t)(s0 + sq);
    const float* x_b    = x + (size_t)b * C_DIM + (size_t)(c0 + sq);

    // ---- software pipeline: preload K-step 0 into registers
    float4 mreg[4], xreg[4];
    #pragma unroll
    for (int rr = 0; rr < 4; ++rr) {
        mreg[rr] = *reinterpret_cast<const float4*>(mask_b + (size_t)(tq + rr) * S_DIM);
        xreg[rr] = *reinterpret_cast<const float4*>(x_b + (size_t)(tq + rr) * (B_DIM * C_DIM));
    }

    for (int kk = 0; kk < T_DIM; kk += 32) {
        // ---- stage current registers to LDS (transpose + hi/lo bf16 split)
        split_store(mreg, sAh, sAl, sq, tq);
        split_store(xreg, sBh, sBl, sq, tq);
        __syncthreads();

        // ---- issue next K-step's global loads now; latency hides behind WMMAs
        if (kk + 32 < T_DIM) {
            const float* mrow = mask_b + (size_t)(kk + 32 + tq) * S_DIM;
            const float* xrow = x_b + (size_t)(kk + 32 + tq) * (B_DIM * C_DIM);
            #pragma unroll
            for (int rr = 0; rr < 4; ++rr) {
                mreg[rr] = *reinterpret_cast<const float4*>(mrow + (size_t)rr * S_DIM);
                xreg[rr] = *reinterpret_cast<const float4*>(xrow + (size_t)rr * (B_DIM * C_DIM));
            }
        }

        // ---- WMMA: ISA A-layout k-base = (lane>=16 ? 8 : 0), runs at kb and kb+16
        //            B uses same k order: k-base = (lane>=16 ? 8 : 0), runs at kb and kb+16
        const int kbA = hi16 * 8;
        const int kbB = hi16 * 8;

        v16bf aH[2], aL[2];
        #pragma unroll
        for (int sb = 0; sb < 2; ++sb) {
            const int row = ((w0 * 2 + sb) * 16 + r) * KPAD;
            aH[sb] = ld_frag(&sAh[row + kbA], &sAh[row + kbA + 16]);
            aL[sb] = ld_frag(&sAl[row + kbA], &sAl[row + kbA + 16]);
        }
        #pragma unroll
        for (int ct = 0; ct < 4; ++ct) {
            const int row = ((w1 * 4 + ct) * 16 + r) * KPAD;
            v16bf bH = ld_frag(&sBh[row + kbB], &sBh[row + kbB + 16]);
            v16bf bL = ld_frag(&sBl[row + kbB], &sBl[row + kbB + 16]);
            #pragma unroll
            for (int sb = 0; sb < 2; ++sb) {
                acc[sb][ct] = __builtin_amdgcn_wmma_f32_16x16x32_bf16(
                    false, aH[sb], false, bH, (short)0, acc[sb][ct], false, false);
                acc[sb][ct] = __builtin_amdgcn_wmma_f32_16x16x32_bf16(
                    false, aH[sb], false, bL, (short)0, acc[sb][ct], false, false);
                acc[sb][ct] = __builtin_amdgcn_wmma_f32_16x16x32_bf16(
                    false, aL[sb], false, bH, (short)0, acc[sb][ct], false, false);
            }
        }
        __syncthreads();  // all fragment reads done before next stage overwrites LDS
    }

    // ---- epilogue: C/D layout -> out[1+s][b][c]
    #pragma unroll
    for (int sb = 0; sb < 2; ++sb) {
        #pragma unroll
        for (int ct = 0; ct < 4; ++ct) {
            const int n = c0 + (w1 * 4 + ct) * 16 + r;
            #pragma unroll
            for (int i = 0; i < 8; ++i) {
                const int m = s0 + (w0 * 2 + sb) * 16 + i + hi16 * 8;
                out[(size_t)(1 + m) * (B_DIM * C_DIM) + (size_t)b * C_DIM + n] = acc[sb][ct][i];
            }
        }
    }
}

// out[0, b, c] = leftmost_group[c]
__global__ void leftmost_fill_kernel(const float* __restrict__ lg, float* __restrict__ out) {
    int idx = blockIdx.x * blockDim.x + threadIdx.x;
    if (idx < B_DIM * C_DIM) out[idx] = lg[idx & (C_DIM - 1)];
}

extern "C" void kernel_launch(void* const* d_in, const int* in_sizes, int n_in,
                              void* d_out, int out_size, void* d_ws, size_t ws_size,
                              hipStream_t stream) {
    const float* x    = (const float*)d_in[0];   // [T,B,C] fp32
    const float* mask = (const float*)d_in[1];   // [B,T,S] fp32
    // d_in[2]: size_of_groups (int64) — unused in 'average' mode reference
    const float* lg   = (const float*)d_in[3];   // [1,1,C] fp32
    float* out = (float*)d_out;                  // [S+1,B,C] fp32

    leftmost_fill_kernel<<<(B_DIM * C_DIM + 255) / 256, 256, 0, stream>>>(lg, out);

    dim3 grid((S_DIM / 128) * (C_DIM / 128), B_DIM);   // (64, 8)
    downsample_gemm_kernel<<<grid, 256, 0, stream>>>(x, mask, out);
}
